// EnhancedCGMNMemory_36558761624498
// MI455X (gfx1250) — matrix operand order repaired and checked by reading, not compiled, not run on hardware
//
#include <hip/hip_runtime.h>
#include <hip/hip_bf16.h>
#include <math.h>

// ---------------------------------------------------------------------------
// Types for CDNA5 WMMA (wave32): v_wmma_f32_16x16x32_bf16
// ---------------------------------------------------------------------------
typedef __attribute__((ext_vector_type(16))) __bf16 v16bf;
typedef __attribute__((ext_vector_type(8)))  __bf16 v8bf;
typedef __attribute__((ext_vector_type(8)))  float  v8f;
typedef __attribute__((ext_vector_type(4)))  float  v4f;

#define NTOK   32768   // 16 * 2048 tokens
#define D3     48
#define D3P    64      // K padded to multiple of 32
#define MSLOT  512
#define HDIM   256
#define TOPK   32
#define INDIM  1024
#define ODEH   128
#define LN_EPS 1e-5f

__device__ __forceinline__ v8f zero8() {
    v8f z;
#pragma unroll
    for (int i = 0; i < 8; ++i) z[i] = 0.0f;
    return z;
}

__device__ __forceinline__ v8f wmma_bf16(v16bf a, v16bf b, v8f c) {
    // D = A(16x32 bf16) * B(32x16 bf16) + C(16x16 f32)
    return __builtin_amdgcn_wmma_f32_16x16x32_bf16(false, a, false, b,
                                                   (short)0, c, false, false);
}

// A fragment from a row-major [16][ldk] bf16 panel. The lane's 16 elements
// form one contiguous 16-element run (two 16B loads):
//   e=0..15 -> K = k0 + 16*half + e
__device__ __forceinline__ v16bf load_a_frag(const __bf16* base, int ldk,
                                             int k0, int lane) {
    const int hi = (lane >> 4) & 1;
    const int m  = lane & 15;
    const __bf16* p = base + m * ldk + k0 + hi * 16;
    const v8bf lo = *(const v8bf*)p;
    const v8bf hh = *(const v8bf*)(p + 8);
    return __builtin_shufflevector(lo, hh, 0, 1, 2, 3, 4, 5, 6, 7, 8, 9, 10,
                                   11, 12, 13, 14, 15);
}

// B fragment from a pre-packed panel: pb[tile][lane][16 bf16 contiguous].
// Packing rule (32Kx16N tile): element e of lane l holds
//   B[k0 + 16*(l>>4) + e][n0 + (l&15)]
__device__ __forceinline__ v16bf load_b_packed(const __bf16* pb, int tile,
                                               int lane) {
    return *(const v16bf*)(pb + ((size_t)tile * 32 + lane) * 16);
}

__device__ __forceinline__ float gelu_exact(float v) {
    return 0.5f * v * (1.0f + erff(v * 0.70710678118654752f));
}

// ---------------------------------------------------------------------------
// Kernel 0a: per-slot ||mem_pos||^2 and curvature weight exp(-alpha*||curv||)
// ---------------------------------------------------------------------------
__global__ void prep_slots_kernel(const float* __restrict__ pos,
                                  const float* __restrict__ curv,
                                  const float* __restrict__ alpha,
                                  float* __restrict__ mm,
                                  float* __restrict__ cw) {
    const int m = blockIdx.x * blockDim.x + threadIdx.x;
    if (m >= MSLOT) return;
    float s = 0.0f;
#pragma unroll
    for (int k = 0; k < D3; ++k) {
        const float p = pos[m * D3 + k];
        s += p * p;
    }
    mm[m] = s;
    float c2 = 0.0f;
#pragma unroll
    for (int k = 0; k < 16; ++k) {
        const float c = curv[m * 16 + k];
        c2 += c * c;
    }
    cw[m] = __expf(-alpha[0] * sqrtf(c2));
}

// ---------------------------------------------------------------------------
// Kernel 0b: pack wo (256x1024 f32) into bf16 B-fragment order:
// wpk[kt(8)][nt(64)][lane(32)][e(16)], K = kt*32 + 16*(lane>>4) + e,
// N = nt*16 + (lane&15)
// ---------------------------------------------------------------------------
__global__ void pack_wo_kernel(const float* __restrict__ wo,
                               __bf16* __restrict__ wpk) {
    const int i = blockIdx.x * blockDim.x + threadIdx.x;
    if (i >= HDIM * INDIM) return;
    const int e  = i & 15;
    const int ln = (i >> 4) & 31;
    const int nt = (i >> 9) & 63;
    const int kt = i >> 15;
    const int k  = kt * 32 + ((ln >> 4) << 4) + e;
    const int n  = nt * 16 + (ln & 15);
    wpk[i] = (__bf16)wo[(size_t)k * INDIM + n];
}

// ---------------------------------------------------------------------------
// Kernel 1: h = gelu(LN(x @ w1 + b1))   (32768x48, K=1024) via bf16 WMMA
// block = 128 threads (4 waves), 64 rows per block, 512 blocks
// ---------------------------------------------------------------------------
__global__ void proj_kernel(const float* __restrict__ x,
                            const float* __restrict__ w1,
                            const float* __restrict__ b1,
                            const float* __restrict__ g1,
                            const float* __restrict__ be1,
                            float* __restrict__ h) {
    __shared__ __align__(32) __bf16 xa[64 * 32];       // 64 rows x 32 K
    __shared__ __align__(32) __bf16 w1pk[3 * 32 * 16]; // packed B tiles
    __shared__ float ds[64 * D3];                      // output staging

    const int tid  = threadIdx.x;
    const int lane = tid & 31;
    const int wv   = tid >> 5;
    const int row0 = blockIdx.x * 64;

    v8f acc[3];
#pragma unroll
    for (int t = 0; t < 3; ++t) acc[t] = zero8();

    for (int k0 = 0; k0 < INDIM; k0 += 32) {
        __syncthreads();
        // stage x tile as bf16 (float4 global loads, streaming NT)
        for (int i = tid; i < 512; i += 128) {
            const int r = i >> 3, c = (i & 7) * 4;
            const v4f v = __builtin_nontemporal_load(
                (const v4f*)&x[(size_t)(row0 + r) * INDIM + k0 + c]);
            __bf16* dst = &xa[r * 32 + c];
            dst[0] = (__bf16)v.x; dst[1] = (__bf16)v.y;
            dst[2] = (__bf16)v.z; dst[3] = (__bf16)v.w;
        }
        // prefetch next x k-tile into cache
        if (k0 + 32 < INDIM) {
            const int r = tid >> 1, c = (tid & 1) * 16;
            __builtin_prefetch(&x[(size_t)(row0 + r) * INDIM + k0 + 32 + c],
                               0, 1);
        }
        // stage w1 slice pre-packed into fragment order
        for (int i = tid; i < 3 * 32 * 16; i += 128) {
            const int e = i & 15, ln = (i >> 4) & 31, t = i >> 9;
            const int k = ((ln >> 4) << 4) + e;
            const int n = t * 16 + (ln & 15);
            w1pk[i] = (__bf16)w1[(size_t)(k0 + k) * D3 + n];
        }
        __syncthreads();
        const v16bf a = load_a_frag(&xa[wv * 16 * 32], 32, 0, lane);
#pragma unroll
        for (int t = 0; t < 3; ++t)
            acc[t] = wmma_bf16(a, load_b_packed(w1pk, t, lane), acc[t]);
    }

    __syncthreads();
    {
        const int hi = lane >> 4, n = lane & 15;
#pragma unroll
        for (int t = 0; t < 3; ++t)
#pragma unroll
            for (int e = 0; e < 8; ++e)
                ds[(wv * 16 + e + 8 * hi) * D3 + t * 16 + n] = acc[t][e];
    }
    __syncthreads();

    if (tid < 64) {
        float buf[D3];
        float s = 0.0f;
#pragma unroll
        for (int c = 0; c < D3; ++c) {
            buf[c] = ds[tid * D3 + c] + b1[c];
            s += buf[c];
        }
        const float mean = s * (1.0f / D3);
        float v = 0.0f;
#pragma unroll
        for (int c = 0; c < D3; ++c) {
            const float d = buf[c] - mean;
            v += d * d;
        }
        const float rs = rsqrtf(v * (1.0f / D3) + LN_EPS);
#pragma unroll
        for (int c = 0; c < D3; ++c) {
            const float y = (buf[c] - mean) * rs * g1[c] + be1[c];
            h[(size_t)(row0 + tid) * D3 + c] = gelu_exact(y);
        }
    }
}

// ---------------------------------------------------------------------------
// Kernel 2: two ODE steps via WMMA.
//   t  = tanh(h @ wa + ba)   (16tok x 128, K=64 zero-padded)
//   dx = t @ wb + bb         (16tok x 48,  K=128)
//   h += 0.5 * dx            (f32 master copy in LDS, bf16 mirror for frags)
// block = 128 threads (4 waves x 16 tokens), 512 blocks, ~65 KB dynamic LDS
// Emits zero-padded bf16 q (K=64) and ||q||^2.
// ---------------------------------------------------------------------------
__global__ void ode_kernel(const float* __restrict__ h_in,
                           const float* __restrict__ wa,
                           const float* __restrict__ ba,
                           const float* __restrict__ wb,
                           const float* __restrict__ bb,
                           __bf16* __restrict__ qb,
                           float* __restrict__ qq) {
    extern __shared__ __align__(32) char smem[];
    __bf16* wapk = (__bf16*)smem;                   // 16 tiles, 16 KB
    __bf16* wbpk = (__bf16*)(smem + 16384);         // 12 tiles, 12 KB
    float*  bas  = (float*)(smem + 28672);          // 128 f32
    float*  bbs  = (float*)(smem + 29184);          // 48 f32 (+pad)
    char*   wbase = smem + 29440;                   // per-wave: 9216 B each
    //   h_bf [16][64]  bf16  (2048 B)
    //   t_bf [16][128] bf16  (4096 B)
    //   h_f  [16][48]  f32   (3072 B)

    const int tid  = threadIdx.x;
    const int lane = tid & 31;
    const int wv   = tid >> 5;
    const int tok0 = blockIdx.x * 64 + wv * 16;

    __bf16* h_bf = (__bf16*)(wbase + wv * 9216);
    __bf16* t_bf = (__bf16*)(wbase + wv * 9216 + 2048);
    float*  h_f  = (float*)(wbase + wv * 9216 + 6144);

    // stage wa pre-packed (tiles nt*2+kt, K zero-padded 48->64)
    for (int i = tid; i < 16 * 32 * 16; i += 128) {
        const int e  = i & 15;
        const int ln = (i >> 4) & 31;
        const int tl = i >> 9;
        const int kt = tl & 1, nt = tl >> 1;
        const int k  = kt * 32 + ((ln >> 4) << 4) + e;
        const int n  = nt * 16 + (ln & 15);
        wapk[i] = (k < D3) ? (__bf16)wa[(size_t)k * ODEH + n] : (__bf16)0.0f;
    }
    // stage wb pre-packed (tiles kt*3+nt2)
    for (int i = tid; i < 12 * 32 * 16; i += 128) {
        const int e  = i & 15;
        const int ln = (i >> 4) & 31;
        const int tl = i >> 9;
        const int kt = tl / 3, nt2 = tl % 3;
        const int k  = kt * 32 + ((ln >> 4) << 4) + e;
        const int n  = nt2 * 16 + (ln & 15);
        wbpk[i] = (__bf16)wb[(size_t)k * D3 + n];
    }
    if (tid < ODEH) bas[tid] = ba[tid];
    if (tid < D3)   bbs[tid] = bb[tid];

    // stage this wave's 16 tokens: f32 master + zero-padded bf16 mirror
    for (int i = lane; i < 16 * D3; i += 32) {
        const int r = i / D3, c = i % D3;
        const float v = h_in[(size_t)(tok0 + r) * D3 + c];
        h_f[i] = v;
        h_bf[r * D3P + c] = (__bf16)v;
    }
    for (int i = lane; i < 16 * 16; i += 32)
        h_bf[(i >> 4) * D3P + D3 + (i & 15)] = (__bf16)0.0f;
    __syncthreads();

    const int hi = lane >> 4, nl = lane & 15;

    for (int step = 0; step < 2; ++step) {
        // t = tanh(h @ wa + ba): K=64 (2 k-tiles), N=128 (8 n-tiles)
        const v16bf a0 = load_a_frag(h_bf, D3P, 0, lane);
        const v16bf a1 = load_a_frag(h_bf, D3P, 32, lane);
#pragma unroll
        for (int nt = 0; nt < 8; ++nt) {
            v8f acc = zero8();
            acc = wmma_bf16(a0, load_b_packed(wapk, nt * 2 + 0, lane), acc);
            acc = wmma_bf16(a1, load_b_packed(wapk, nt * 2 + 1, lane), acc);
            const int n = nt * 16 + nl;
            const float bv = bas[n];
#pragma unroll
            for (int e = 0; e < 8; ++e) {
                const int m = e + 8 * hi;
                t_bf[m * ODEH + n] = (__bf16)tanhf(acc[e] + bv);
            }
        }
        // dx = t @ wb + bb: K=128 (4 k-tiles), N=48 (3 n-tiles)
        v8f accd[3];
#pragma unroll
        for (int t = 0; t < 3; ++t) accd[t] = zero8();
#pragma unroll
        for (int kt = 0; kt < 4; ++kt) {
            const v16bf a = load_a_frag(t_bf, ODEH, kt * 32, lane);
#pragma unroll
            for (int t = 0; t < 3; ++t)
                accd[t] =
                    wmma_bf16(a, load_b_packed(wbpk, kt * 3 + t, lane), accd[t]);
        }
        // h += 0.5 * (dx + bb); refresh bf16 mirror
#pragma unroll
        for (int t = 0; t < 3; ++t) {
            const int j = t * 16 + nl;
            const float bv = bbs[j];
#pragma unroll
            for (int e = 0; e < 8; ++e) {
                const int m = e + 8 * hi;
                const float hn = h_f[m * D3 + j] + 0.5f * (accd[t][e] + bv);
                h_f[m * D3 + j] = hn;
                h_bf[m * D3P + j] = (__bf16)hn;
            }
        }
    }

    // emit q (bf16, padded) and ||q||^2 (f32, from the f32 master copy)
    {
        const uint4* src = (const uint4*)h_bf;
        uint4* dst = (uint4*)(qb + (size_t)tok0 * D3P);
        for (int i = lane; i < 16 * D3P / 8; i += 32) dst[i] = src[i];
    }
    if (lane < 16) {
        float s = 0.0f;
#pragma unroll
        for (int c = 0; c < D3; ++c) {
            const float v = h_f[lane * D3 + c];
            s += v * v;
        }
        qq[tok0 + lane] = s;
    }
}

// ---------------------------------------------------------------------------
// Kernel 3: fused dist (WMMA) + top-32 + softmax + mem gather -> attended bf16
// block = 128 threads (4 waves x 16 tokens), 512 blocks, ~204 KB dynamic LDS
// mem_pos^T staged pre-packed in fragment order: pb[nt*2+kt][lane][16]
// ---------------------------------------------------------------------------
__global__ void attend_kernel(const __bf16* __restrict__ qb,
                              const float* __restrict__ qq,
                              const float* __restrict__ pos,
                              const float* __restrict__ mem,
                              const float* __restrict__ mm,
                              const float* __restrict__ cw,
                              __bf16* __restrict__ att) {
    extern __shared__ __align__(32) char smem[];
    __bf16* pb    = (__bf16*)smem;                           // 64 tiles, 64 KB
    float*  mms   = (float*)(smem + 65536);                  // 2 KB
    float*  cws   = (float*)(smem + 65536 + 2048);           // 2 KB
    __bf16* qs    = (__bf16*)(smem + 65536 + 4096);          // 4*[16][64] 8 KB
    float*  distL = (float*)(smem + 65536 + 4096 + 8192);    // 4*[16][512]

    const int tid  = threadIdx.x;
    const int lane = tid & 31;
    const int wv   = tid >> 5;
    const int tok0 = blockIdx.x * 64 + wv * 16;

    // stage mem_pos^T pre-packed into B-fragment order (zero-padded K)
    for (int i = tid; i < D3P * MSLOT; i += 128) {
        const int e  = i & 15;
        const int ln = (i >> 4) & 31;
        const int tl = i >> 9;          // tile = nt*2 + kt
        const int kt = tl & 1, nt = tl >> 1;
        const int k  = kt * 32 + ((ln >> 4) << 4) + e;
        const int n  = nt * 16 + (ln & 15);
        pb[i] = (k < D3) ? (__bf16)pos[(size_t)n * D3 + k] : (__bf16)0.0f;
    }
    for (int i = tid; i < MSLOT; i += 128) { mms[i] = mm[i]; cws[i] = cw[i]; }
    // stage this wave's 16 query rows (padded K=64) via 16B copies
    __bf16* myq = qs + wv * 16 * D3P;
    {
        const uint4* qsrc = (const uint4*)(qb + (size_t)tok0 * D3P);
        uint4* qdst = (uint4*)myq;
        for (int i = lane; i < 16 * D3P / 8; i += 32) qdst[i] = qsrc[i];
    }
    __syncthreads();

    const v16bf a0 = load_a_frag(myq, D3P, 0, lane);
    const v16bf a1 = load_a_frag(myq, D3P, 32, lane);
    float* myd = distL + wv * 16 * MSLOT;

    const int hi = lane >> 4, nl = lane & 15;
    float qv[8];
#pragma unroll
    for (int e = 0; e < 8; ++e) qv[e] = qq[tok0 + e + 8 * hi];

    for (int nt = 0; nt < MSLOT / 16; ++nt) {
        v8f acc = zero8();
        acc = wmma_bf16(a0, load_b_packed(pb, nt * 2 + 0, lane), acc);
        acc = wmma_bf16(a1, load_b_packed(pb, nt * 2 + 1, lane), acc);
        const int slot = nt * 16 + nl;
        const float mmv = mms[slot], cwv = cws[slot];
#pragma unroll
        for (int e = 0; e < 8; ++e) {
            const int m = e + 8 * hi;
            float d = qv[e] - 2.0f * acc[e] + mmv;
            d = fmaxf(d, 0.0f) * cwv;
            myd[m * MSLOT + slot] = d;
        }
    }

    // per-token wave-cooperative top-32 selection + softmax + gather
    for (int t = 0; t < 16; ++t) {
        const float* dptr = myd + t * MSLOT;
        float dv[16];
#pragma unroll
        for (int j = 0; j < 16; ++j) dv[j] = dptr[lane + 32 * j];

        float selV = 0.0f;
        int   selI = 0;
        for (int k = 0; k < TOPK; ++k) {
            float mv = 3.4e38f;
            int   mj = 0;
#pragma unroll
            for (int j = 0; j < 16; ++j)
                if (dv[j] < mv) { mv = dv[j]; mj = j; }
            int gi = lane + (mj << 5);
#pragma unroll
            for (int o = 16; o > 0; o >>= 1) {
                const float ov = __shfl_xor(mv, o, 32);
                const int   oi = __shfl_xor(gi, o, 32);
                if (ov < mv || (ov == mv && oi < gi)) { mv = ov; gi = oi; }
            }
            if (lane == k) { selV = mv; selI = gi; }
            if ((gi & 31) == lane) dv[gi >> 5] = 3.4e38f;
        }

        // softmax(-selected) across the 32 lanes
        float nv = -selV;
        float mx = nv;
#pragma unroll
        for (int o = 16; o > 0; o >>= 1) mx = fmaxf(mx, __shfl_xor(mx, o, 32));
        const float ev = __expf(nv - mx);
        float sm = ev;
#pragma unroll
        for (int o = 16; o > 0; o >>= 1) sm += __shfl_xor(sm, o, 32);
        const float wgt = ev / sm;

        // attended = sum_k w_k * mem[idx_k]; lane owns 8 of 256 channels
        float acc8[8];
#pragma unroll
        for (int c = 0; c < 8; ++c) acc8[c] = 0.0f;
        for (int k = 0; k < TOPK; ++k) {
            const float wk = __shfl(wgt, k, 32);
            const int   ik = __shfl(selI, k, 32);
            const float4* rp =
                (const float4*)(mem + (size_t)ik * HDIM + lane * 8);
            const float4 u = rp[0], v = rp[1];
            acc8[0] += wk * u.x; acc8[1] += wk * u.y;
            acc8[2] += wk * u.z; acc8[3] += wk * u.w;
            acc8[4] += wk * v.x; acc8[5] += wk * v.y;
            acc8[6] += wk * v.z; acc8[7] += wk * v.w;
        }
        __bf16* ap = att + (size_t)(tok0 + t) * HDIM + lane * 8;
#pragma unroll
        for (int c = 0; c < 8; ++c) ap[c] = (__bf16)acc8[c];
    }
}

// ---------------------------------------------------------------------------
// Kernel 4: out = gelu(LN(att @ wo + bo))   (32768x1024, K=256) via bf16 WMMA
// block = 256 threads (8 waves), 32 rows per block, 1024 blocks.
// B fragments read directly from pre-packed global wo (coalesced b128, L2-hot)
// ---------------------------------------------------------------------------
__global__ void outp_kernel(const __bf16* __restrict__ att,
                            const __bf16* __restrict__ wpk,
                            const float* __restrict__ bo,
                            const float* __restrict__ go,
                            const float* __restrict__ beo,
                            float* __restrict__ out) {
    extern __shared__ __align__(32) char smem[];
    __bf16* att_s = (__bf16*)smem;              // [32][256] 16 KB
    float*  outs  = (float*)(smem + 16384);     // [32][1024] 128 KB

    const int tid  = threadIdx.x;
    const int lane = tid & 31;
    const int wv   = tid >> 5;
    const int rg   = wv & 1;       // row group (16 rows)
    const int cg   = wv >> 1;      // column group (256 cols = 16 tiles)
    const int row0 = blockIdx.x * 32;

    {
        const uint4* src = (const uint4*)(att + (size_t)row0 * HDIM);
        uint4* dst = (uint4*)att_s;
        for (int i = tid; i < 32 * HDIM / 8; i += 256) dst[i] = src[i];
    }
    __syncthreads();

    v8f acc[16];
#pragma unroll
    for (int t = 0; t < 16; ++t) acc[t] = zero8();

    for (int kt = 0; kt < HDIM / 32; ++kt) {
        const v16bf a = load_a_frag(att_s + rg * 16 * HDIM, HDIM, kt * 32, lane);
#pragma unroll
        for (int t = 0; t < 16; ++t)
            acc[t] = wmma_bf16(
                a, load_b_packed(wpk, kt * 64 + cg * 16 + t, lane), acc[t]);
    }

    {
        const int hi = lane >> 4, nl = lane & 15;
#pragma unroll
        for (int t = 0; t < 16; ++t)
#pragma unroll
            for (int e = 0; e < 8; ++e)
                outs[(rg * 16 + e + 8 * hi) * INDIM + cg * 256 + t * 16 + nl] =
                    acc[t][e];
    }
    __syncthreads();

    // row-wise LN(1024) + GELU: one wave per row, lane owns 32 columns
    for (int rr = wv; rr < 32; rr += 8) {
        float vbuf[32];
        float s = 0.0f, s2 = 0.0f;
#pragma unroll
        for (int i = 0; i < 32; ++i) {
            const int j = lane + 32 * i;
            const float v = outs[rr * INDIM + j] + bo[j];
            vbuf[i] = v; s += v; s2 += v * v;
        }
#pragma unroll
        for (int o = 16; o > 0; o >>= 1) {
            s  += __shfl_xor(s, o, 32);
            s2 += __shfl_xor(s2, o, 32);
        }
        const float mean = s * (1.0f / INDIM);
        const float var  = s2 * (1.0f / INDIM) - mean * mean;
        const float rs   = rsqrtf(var + LN_EPS);
#pragma unroll
        for (int i = 0; i < 32; ++i) {
            const int j = lane + 32 * i;
            const float y = (vbuf[i] - mean) * rs * go[j] + beo[j];
            __builtin_nontemporal_store(
                gelu_exact(y), &out[(size_t)(row0 + rr) * INDIM + j]);
        }
    }
}

// ---------------------------------------------------------------------------
extern "C" void kernel_launch(void* const* d_in, const int* in_sizes, int n_in,
                              void* d_out, int out_size, void* d_ws,
                              size_t ws_size, hipStream_t stream) {
    (void)in_sizes; (void)n_in; (void)out_size; (void)ws_size;
    const float* x    = (const float*)d_in[0];
    const float* w1   = (const float*)d_in[1];
    const float* b1   = (const float*)d_in[2];
    const float* g1   = (const float*)d_in[3];
    const float* be1  = (const float*)d_in[4];
    const float* wa   = (const float*)d_in[5];
    const float* ba   = (const float*)d_in[6];
    const float* wb   = (const float*)d_in[7];
    const float* bb   = (const float*)d_in[8];
    const float* mem  = (const float*)d_in[9];
    const float* pos  = (const float*)d_in[10];
    const float* curv = (const float*)d_in[11];
    const float* alph = (const float*)d_in[12];
    const float* wo   = (const float*)d_in[13];
    const float* bo   = (const float*)d_in[14];
    const float* go   = (const float*)d_in[15];
    const float* beo  = (const float*)d_in[16];
    float* out = (float*)d_out;

    char* ws = (char*)d_ws;
    size_t off = 0;
    auto carve = [&](size_t bytes) {
        char* p = ws + off;
        off = (off + bytes + 255) & ~(size_t)255;
        return p;
    };
    float*  h_buf = (float*)carve((size_t)NTOK * D3 * 4);        //  6 MB
    __bf16* q_bf  = (__bf16*)carve((size_t)NTOK * D3P * 2);      //  4 MB
    float*  qq    = (float*)carve((size_t)NTOK * 4);             // 128 KB
    float*  mm    = (float*)carve((size_t)MSLOT * 4);
    float*  cw    = (float*)carve((size_t)MSLOT * 4);
    __bf16* att   = (__bf16*)carve((size_t)NTOK * HDIM * 2);     // 16 MB
    __bf16* wpk   = (__bf16*)carve((size_t)HDIM * INDIM * 2);    // 512 KB

    prep_slots_kernel<<<2, 256, 0, stream>>>(pos, curv, alph, mm, cw);
    pack_wo_kernel<<<(HDIM * INDIM + 255) / 256, 256, 0, stream>>>(wo, wpk);
    proj_kernel<<<NTOK / 64, 128, 0, stream>>>(x, w1, b1, g1, be1, h_buf);
    ode_kernel<<<NTOK / 64, 128, 66304, stream>>>(h_buf, wa, ba, wb, bb, q_bf,
                                                  qq);
    attend_kernel<<<NTOK / 64, 128, 208896, stream>>>(q_bf, qq, pos, mem, mm,
                                                      cw, att);
    outp_kernel<<<NTOK / 32, 256, 147456, stream>>>(att, wpk, bo, go, beo, out);
}
